// GRU_GAT_Model_78640851189808
// MI455X (gfx1250) — matrix operand order repaired and checked by reading, not compiled, not run on hardware
//
#include <hip/hip_runtime.h>
#include <hip/hip_bf16.h>

// ---------------------------------------------------------------------------
// Types for CDNA5 WMMA (wave32): V_WMMA_F32_16X16X32_BF16
// ---------------------------------------------------------------------------
typedef __attribute__((ext_vector_type(16))) __bf16 bf16x16;
typedef __attribute__((ext_vector_type(8)))  float  f32x8;

#define N_NODES 10000
#define N_EDGES 320000
#define IN_DIM  64
#define HID     128
#define HEADS   4
#define OUT_DIM 64
#define G3      (3 * HID)      // 384
#define HC1     (HEADS * HID)  // 512

// ---------------------------------------------------------------------------
// fp32 -> bf16 (round-to-nearest-even)
// ---------------------------------------------------------------------------
__device__ __forceinline__ unsigned short f2bf(float f) {
    unsigned int u = __float_as_uint(f);
    u += 0x7FFFu + ((u >> 16) & 1u);
    return (unsigned short)(u >> 16);
}

__global__ void pack_bf16(const float* __restrict__ x,
                          unsigned short* __restrict__ a, long long n) {
    long long i = (long long)blockIdx.x * blockDim.x + threadIdx.x;
    if (i < n) a[i] = f2bf(x[i]);
}

// ---------------------------------------------------------------------------
// Pack B into WMMA-fragment-major bf16:
//   Bp[(tn*Ktiles + kt)*512 + lane*16 + e]
// with (per ISA 7.12.2, wave32) lane = g*16 + nl,  n = tn*16+nl,
//   k = kt*32 + ((e&8)<<1) + g*8 + (e&7).
// transposed=1: source w is [Nc,K] row-major (use w[n*K+k], builds W^T)
// transposed=0: source w is [K,Nc] row-major (use w[k*Nc+n])
// ---------------------------------------------------------------------------
__global__ void pack_bf16_frag(const float* __restrict__ w,
                               unsigned short* __restrict__ Bp,
                               int K, int Nc, int transposed) {
    int i = blockIdx.x * blockDim.x + threadIdx.x;
    if (i >= K * Nc) return;
    const int ktiles = K >> 5;
    int e    = i & 15;
    int lane = (i >> 4) & 31;
    int tile = i >> 9;
    int tn = tile / ktiles, kt = tile - tn * ktiles;
    int g = lane >> 4, nl = lane & 15;
    int n = tn * 16 + nl;
    int k = kt * 32 + ((e & 8) << 1) + g * 8 + (e & 7);
    float v = transposed ? w[(size_t)n * K + k] : w[(size_t)k * Nc + n];
    Bp[i] = f2bf(v);
}

__global__ void fill_u32(unsigned int* __restrict__ p, unsigned int v, long long n) {
    long long i = (long long)blockIdx.x * blockDim.x + threadIdx.x;
    if (i < n) p[i] = v;
}

// ---------------------------------------------------------------------------
// WMMA GEMM: C[M,Nc] = A[M,K](bf16) * B(frag-packed bf16) (+ bias), fp32 out.
// One wave computes a 16x64 C strip (4 N-tiles): one A fragment feeds 4
// back-to-back v_wmma per K step. All fragments are contiguous b128 loads.
// Layouts per CDNA5 ISA 7.12.2 (wave32):
//   A 16x32: lane L (g=L/16, m=L%16): e<8 -> k=g*8+e ; e>=8 -> k=16+g*8+(e-8)
//   B 32x16: lane L (g=L/16, n=L%16): e<8 -> k=g*8+e ; e>=8 -> k=16+g*8+(e-8) (pre-swizzled)
//   C 16x16: lane L (g=L/16, n=L%16): vgpr v -> row m=g*8+v
// ---------------------------------------------------------------------------
__global__ __launch_bounds__(256) void wmma_gemm_bf16(
    const unsigned short* __restrict__ A,
    const unsigned short* __restrict__ Bp,
    const float* __restrict__ bias,
    float* __restrict__ C,
    int M, int K, int Nc)
{
    const int lane = threadIdx.x & 31;
    const int wave = threadIdx.x >> 5;
    const int tilesN4 = Nc >> 6;          // groups of 4 N-tiles
    const int tilesM  = M >> 4;
    int grp = blockIdx.x * 8 + wave;
    if (grp >= tilesM * tilesN4) return;
    const int tm  = grp / tilesN4;
    const int tn4 = grp - tm * tilesN4;
    const int g = lane >> 4;
    const int r = lane & 15;
    const int ktiles = K >> 5;

    f32x8 acc0 = {}, acc1 = {}, acc2 = {}, acc3 = {};
    const unsigned short* Arow = A + (size_t)(tm * 16 + r) * K;
    // Per-lane fragment base for the 4 consecutive N tiles
    const unsigned short* Bb = Bp + ((size_t)(tn4 * 4) * ktiles) * 512 + lane * 16;

    for (int kt = 0; kt < ktiles; ++kt) {
        __builtin_prefetch(Arow + kt * 32 + 32, 0, 1);   // global_prefetch_b8
        union { uint4 u[2]; bf16x16 v; } au;
        au.u[0] = *(const uint4*)(Arow + kt * 32 + g * 8);
        au.u[1] = *(const uint4*)(Arow + kt * 32 + 16 + g * 8);

        union { uint4 u[2]; bf16x16 v; } b0, b1, b2, b3;
        const unsigned short* bp0 = Bb + (size_t)kt * 512;
        const unsigned short* bp1 = bp0 + (size_t)ktiles * 512;
        const unsigned short* bp2 = bp1 + (size_t)ktiles * 512;
        const unsigned short* bp3 = bp2 + (size_t)ktiles * 512;
        b0.u[0] = *(const uint4*)(bp0);  b0.u[1] = *(const uint4*)(bp0 + 8);
        b1.u[0] = *(const uint4*)(bp1);  b1.u[1] = *(const uint4*)(bp1 + 8);
        b2.u[0] = *(const uint4*)(bp2);  b2.u[1] = *(const uint4*)(bp2 + 8);
        b3.u[0] = *(const uint4*)(bp3);  b3.u[1] = *(const uint4*)(bp3 + 8);

        acc0 = __builtin_amdgcn_wmma_f32_16x16x32_bf16(false, au.v, false, b0.v, (short)0, acc0, false, false);
        acc1 = __builtin_amdgcn_wmma_f32_16x16x32_bf16(false, au.v, false, b1.v, (short)0, acc1, false, false);
        acc2 = __builtin_amdgcn_wmma_f32_16x16x32_bf16(false, au.v, false, b2.v, (short)0, acc2, false, false);
        acc3 = __builtin_amdgcn_wmma_f32_16x16x32_bf16(false, au.v, false, b3.v, (short)0, acc3, false, false);
    }

    // Store 4 tiles: columns tn4*64 .. tn4*64+63
    const int ncol0 = tn4 * 64 + r;
    float* Crow = C + (size_t)(tm * 16 + g * 8) * Nc;
    f32x8 av[4] = { acc0, acc1, acc2, acc3 };
#pragma unroll
    for (int j = 0; j < 4; ++j) {
        const int n = ncol0 + j * 16;
        const float bn = bias ? bias[n] : 0.0f;
        float* Cp = Crow + n;
#pragma unroll
        for (int v = 0; v < 8; ++v) Cp[(size_t)v * Nc] = av[j][v] + bn;
    }
}

// ---------------------------------------------------------------------------
// Sequential GRU scan. One workgroup, 384 threads, thread t owns row t of
// w_hh (128 fp32 kept in VGPRs for all T steps). h broadcast from LDS.
// gi = x @ w_ih^T + b_ih precomputed by WMMA GEMM. Gates [r,z,n].
// ---------------------------------------------------------------------------
__device__ __forceinline__ float sigm(float x) { return 1.0f / (1.0f + __expf(-x)); }

__global__ __launch_bounds__(384) void gru_scan(
    const float* __restrict__ gi,    // [T, 384]
    const float* __restrict__ w_hh,  // [384, 128]
    const float* __restrict__ b_hh,  // [384]
    float* __restrict__ hout,        // [T, 128]
    int T)
{
    __shared__ float h[HID];
    __shared__ float gh[G3];
    const int t = threadIdx.x;

    float w[HID];
#pragma unroll
    for (int i = 0; i < HID / 4; ++i) {
        float4 v = *(const float4*)(w_hh + (size_t)t * HID + 4 * i);
        w[4 * i] = v.x; w[4 * i + 1] = v.y; w[4 * i + 2] = v.z; w[4 * i + 3] = v.w;
    }
    const float bh = b_hh[t];
    if (t < HID) h[t] = 0.0f;
    __syncthreads();

    for (int s = 0; s < T; ++s) {
        float acc = bh;
#pragma unroll
        for (int i = 0; i < HID / 4; ++i) {
            float4 hv = *(const float4*)(h + 4 * i);
            acc = fmaf(w[4 * i],     hv.x, acc);
            acc = fmaf(w[4 * i + 1], hv.y, acc);
            acc = fmaf(w[4 * i + 2], hv.z, acc);
            acc = fmaf(w[4 * i + 3], hv.w, acc);
        }
        gh[t] = acc;
        __syncthreads();
        if (t < HID) {
            const float* git = gi + (size_t)s * G3;
            float r = sigm(git[t] + gh[t]);
            float z = sigm(git[HID + t] + gh[HID + t]);
            float n = tanhf(git[2 * HID + t] + r * gh[2 * HID + t]);
            float hn = (1.0f - z) * n + z * h[t];
            h[t] = hn;
            hout[(size_t)s * HID + t] = hn;
        }
        __syncthreads();
    }
}

// ---------------------------------------------------------------------------
// GAT attention scores: a_s[n,h] = <hp[n,h,:], att_src[h,:]> (and a_d).
// One wave32 per (node, head); shuffle reduction.
// ---------------------------------------------------------------------------
__global__ void att_scores(const float* __restrict__ hp,
                           const float* __restrict__ att_s,
                           const float* __restrict__ att_d,
                           float* __restrict__ as_, float* __restrict__ ad_,
                           int N, int nheads, int C)
{
    long long wv = ((long long)blockIdx.x * blockDim.x + threadIdx.x) >> 5;
    int lane = threadIdx.x & 31;
    long long total = (long long)N * nheads;
    if (wv >= total) return;
    int n = (int)(wv / nheads), hd = (int)(wv % nheads);
    const float* hrow = hp + (size_t)n * nheads * C + (size_t)hd * C;
    const float* vs = att_s + hd * C;
    const float* vd = att_d + hd * C;
    float ss = 0.0f, sd = 0.0f;
    for (int c = lane; c < C; c += 32) {
        float hv = hrow[c];
        ss = fmaf(hv, vs[c], ss);
        sd = fmaf(hv, vd[c], sd);
    }
    for (int off = 16; off; off >>= 1) {
        ss += __shfl_down(ss, off, 32);
        sd += __shfl_down(sd, off, 32);
    }
    if (lane == 0) { as_[wv] = ss; ad_[wv] = sd; }
}

// ---------------------------------------------------------------------------
// Edge softmax pass 1: per-dst running max (sign-split int/uint atomic trick,
// buffer initialized to -inf bits 0xFF800000).
// ---------------------------------------------------------------------------
__device__ __forceinline__ void atomicMaxF(float* addr, float val) {
    if (val >= 0.0f) atomicMax((int*)addr, __float_as_int(val));
    else             atomicMin((unsigned int*)addr, __float_as_uint(val));
}

__global__ void edge_softmax_max(const long long* __restrict__ src,
                                 const long long* __restrict__ dst,
                                 const float* __restrict__ as_,
                                 const float* __restrict__ ad_,
                                 float* __restrict__ m,
                                 int E, int N, int nheads)
{
    long long idx = (long long)blockIdx.x * blockDim.x + threadIdx.x;
    long long total = (long long)(E + N) * nheads;
    if (idx >= total) return;
    int e = (int)(idx / nheads), hd = (int)(idx % nheads);
    int si, di;
    if (e < E) { si = (int)src[e]; di = (int)dst[e]; }
    else       { si = di = e - E; }           // self loops appended
    float v = as_[(size_t)si * nheads + hd] + ad_[(size_t)di * nheads + hd];
    v = v > 0.0f ? v : 0.2f * v;              // leaky_relu(0.2)
    atomicMaxF(m + (size_t)di * nheads + hd, v);
}

// ---------------------------------------------------------------------------
// Edge softmax pass 2: ex = exp(e - m[d]); denom[d] += ex; acc[d] += ex*h[s].
// One wave32 per (edge, head); channels lane-parallel.
// ---------------------------------------------------------------------------
__global__ void edge_softmax_acc(const long long* __restrict__ src,
                                 const long long* __restrict__ dst,
                                 const float* __restrict__ as_,
                                 const float* __restrict__ ad_,
                                 const float* __restrict__ m,
                                 const float* __restrict__ hp,
                                 float* __restrict__ denom,
                                 float* __restrict__ acc,
                                 int E, int N, int nheads, int C)
{
    long long wv = ((long long)blockIdx.x * blockDim.x + threadIdx.x) >> 5;
    int lane = threadIdx.x & 31;
    long long total = (long long)(E + N) * nheads;
    if (wv >= total) return;
    int e = (int)(wv / nheads), hd = (int)(wv % nheads);
    int si, di;
    if (e < E) { si = (int)src[e]; di = (int)dst[e]; }
    else       { si = di = e - E; }
    float v = as_[(size_t)si * nheads + hd] + ad_[(size_t)di * nheads + hd];
    v = v > 0.0f ? v : 0.2f * v;
    float ex = __expf(v - m[(size_t)di * nheads + hd]);
    if (lane == 0) atomicAdd(denom + (size_t)di * nheads + hd, ex);
    const float* hrow = hp + (size_t)si * nheads * C + (size_t)hd * C;
    float* arow = acc + (size_t)di * nheads * C + (size_t)hd * C;
    for (int c = lane; c < C; c += 32) atomicAdd(arow + c, ex * hrow[c]);
}

// In-place: acc[i] = (acc[i]/denom + bias) (optionally relu)
__global__ void gat_finalize(float* __restrict__ acc,
                             const float* __restrict__ denom,
                             const float* __restrict__ bias,
                             int N, int nheads, int C, int do_relu)
{
    long long i = (long long)blockIdx.x * blockDim.x + threadIdx.x;
    long long total = (long long)N * nheads * C;
    if (i >= total) return;
    int rem = (int)(i % (nheads * C));
    int n = (int)(i / (nheads * C));
    int hd = rem / C;
    float v = acc[i] / denom[(size_t)n * nheads + hd] + bias[rem];
    if (do_relu) v = fmaxf(v, 0.0f);
    acc[i] = v;
}

// ---------------------------------------------------------------------------
// Host orchestration
// ---------------------------------------------------------------------------
static inline unsigned int cdiv(long long a, long long b) {
    return (unsigned int)((a + b - 1) / b);
}

extern "C" void kernel_launch(void* const* d_in, const int* in_sizes, int n_in,
                              void* d_out, int out_size, void* d_ws, size_t ws_size,
                              hipStream_t stream) {
    const float* x       = (const float*)d_in[0];
    const long long* ei  = (const long long*)d_in[1];
    const float* w_ih0   = (const float*)d_in[2];
    const float* w_hh0   = (const float*)d_in[3];
    const float* b_ih0   = (const float*)d_in[4];
    const float* b_hh0   = (const float*)d_in[5];
    const float* w_ih1   = (const float*)d_in[6];
    const float* w_hh1   = (const float*)d_in[7];
    const float* b_ih1   = (const float*)d_in[8];
    const float* b_hh1   = (const float*)d_in[9];
    const float* W1      = (const float*)d_in[10];
    const float* att_s1  = (const float*)d_in[11];
    const float* att_d1  = (const float*)d_in[12];
    const float* bias1   = (const float*)d_in[13];
    const float* W2      = (const float*)d_in[14];
    const float* att_s2  = (const float*)d_in[15];
    const float* att_d2  = (const float*)d_in[16];
    const float* bias2   = (const float*)d_in[17];

    const long long* src = ei;
    const long long* dst = ei + N_EDGES;

    // Workspace carve-out (256B aligned)
    char* ws = (char*)d_ws;
    size_t off = 0;
    auto alloc = [&](size_t bytes) -> char* {
        off = (off + 255) & ~(size_t)255;
        char* p = ws + off;
        off += bytes;
        return p;
    };
    unsigned short* bufA  = (unsigned short*)alloc((size_t)N_NODES * HC1 * 2);  // max A: 10000x512 bf16
    unsigned short* bufB  = (unsigned short*)alloc((size_t)HID * HC1 * 2);      // max B: 128x512 bf16
    float* GI   = (float*)alloc((size_t)N_NODES * G3  * 4);
    float* H0   = (float*)alloc((size_t)N_NODES * HID * 4);
    float* H1   = (float*)alloc((size_t)N_NODES * HID * 4);
    float* HP1  = (float*)alloc((size_t)N_NODES * HC1 * 4);
    float* ACC1 = (float*)alloc((size_t)N_NODES * HC1 * 4);                     // becomes X2 in-place
    float* HP2  = (float*)alloc((size_t)N_NODES * OUT_DIM * 4);
    float* AS1  = (float*)alloc((size_t)N_NODES * HEADS * 4);
    float* AD1  = (float*)alloc((size_t)N_NODES * HEADS * 4);
    float* M1   = (float*)alloc((size_t)N_NODES * HEADS * 4);
    float* DN1  = (float*)alloc((size_t)N_NODES * HEADS * 4);
    float* AS2  = (float*)alloc((size_t)N_NODES * 4);
    float* AD2  = (float*)alloc((size_t)N_NODES * 4);
    float* M2   = (float*)alloc((size_t)N_NODES * 4);
    float* DN2  = (float*)alloc((size_t)N_NODES * 4);

    const unsigned int NEG_INF = 0xFF800000u;

    // ---------------- GRU layer 0 ----------------
    pack_bf16<<<cdiv((long long)N_NODES * IN_DIM, 256), 256, 0, stream>>>(
        x, bufA, (long long)N_NODES * IN_DIM);
    pack_bf16_frag<<<cdiv(IN_DIM * G3, 256), 256, 0, stream>>>(
        w_ih0, bufB, IN_DIM, G3, /*transposed=*/1);
    wmma_gemm_bf16<<<cdiv((N_NODES / 16) * (G3 / 64), 8), 256, 0, stream>>>(
        bufA, bufB, b_ih0, GI, N_NODES, IN_DIM, G3);
    gru_scan<<<1, G3, 0, stream>>>(GI, w_hh0, b_hh0, H0, N_NODES);

    // ---------------- GRU layer 1 ----------------
    pack_bf16<<<cdiv((long long)N_NODES * HID, 256), 256, 0, stream>>>(
        H0, bufA, (long long)N_NODES * HID);
    pack_bf16_frag<<<cdiv(HID * G3, 256), 256, 0, stream>>>(
        w_ih1, bufB, HID, G3, /*transposed=*/1);
    wmma_gemm_bf16<<<cdiv((N_NODES / 16) * (G3 / 64), 8), 256, 0, stream>>>(
        bufA, bufB, b_ih1, GI, N_NODES, HID, G3);
    gru_scan<<<1, G3, 0, stream>>>(GI, w_hh1, b_hh1, H1, N_NODES);

    // ---------------- GAT layer 1 ----------------
    pack_bf16<<<cdiv((long long)N_NODES * HID, 256), 256, 0, stream>>>(
        H1, bufA, (long long)N_NODES * HID);
    pack_bf16_frag<<<cdiv(HID * HC1, 256), 256, 0, stream>>>(
        W1, bufB, HID, HC1, /*transposed=*/0);
    wmma_gemm_bf16<<<cdiv((N_NODES / 16) * (HC1 / 64), 8), 256, 0, stream>>>(
        bufA, bufB, nullptr, HP1, N_NODES, HID, HC1);

    att_scores<<<cdiv((long long)N_NODES * HEADS * 32, 256), 256, 0, stream>>>(
        HP1, att_s1, att_d1, AS1, AD1, N_NODES, HEADS, HID);

    fill_u32<<<cdiv((long long)N_NODES * HEADS, 256), 256, 0, stream>>>(
        (unsigned int*)M1, NEG_INF, (long long)N_NODES * HEADS);
    fill_u32<<<cdiv((long long)N_NODES * HEADS, 256), 256, 0, stream>>>(
        (unsigned int*)DN1, 0u, (long long)N_NODES * HEADS);
    fill_u32<<<cdiv((long long)N_NODES * HC1, 256), 256, 0, stream>>>(
        (unsigned int*)ACC1, 0u, (long long)N_NODES * HC1);

    edge_softmax_max<<<cdiv((long long)(N_EDGES + N_NODES) * HEADS, 256), 256, 0, stream>>>(
        src, dst, AS1, AD1, M1, N_EDGES, N_NODES, HEADS);
    edge_softmax_acc<<<cdiv((long long)(N_EDGES + N_NODES) * HEADS * 32, 256), 256, 0, stream>>>(
        src, dst, AS1, AD1, M1, HP1, DN1, ACC1, N_EDGES, N_NODES, HEADS, HID);
    gat_finalize<<<cdiv((long long)N_NODES * HC1, 256), 256, 0, stream>>>(
        ACC1, DN1, bias1, N_NODES, HEADS, HID, /*relu=*/1);

    // ---------------- GAT layer 2 ----------------
    pack_bf16<<<cdiv((long long)N_NODES * HC1, 256), 256, 0, stream>>>(
        ACC1, bufA, (long long)N_NODES * HC1);
    pack_bf16_frag<<<cdiv(HC1 * OUT_DIM, 256), 256, 0, stream>>>(
        W2, bufB, HC1, OUT_DIM, /*transposed=*/0);
    wmma_gemm_bf16<<<cdiv((N_NODES / 16) * (OUT_DIM / 64), 8), 256, 0, stream>>>(
        bufA, bufB, nullptr, HP2, N_NODES, HC1, OUT_DIM);

    att_scores<<<cdiv((long long)N_NODES * 32, 256), 256, 0, stream>>>(
        HP2, att_s2, att_d2, AS2, AD2, N_NODES, 1, OUT_DIM);

    fill_u32<<<cdiv((long long)N_NODES, 256), 256, 0, stream>>>(
        (unsigned int*)M2, NEG_INF, (long long)N_NODES);
    fill_u32<<<cdiv((long long)N_NODES, 256), 256, 0, stream>>>(
        (unsigned int*)DN2, 0u, (long long)N_NODES);
    fill_u32<<<cdiv((long long)N_NODES * OUT_DIM, 256), 256, 0, stream>>>(
        (unsigned int*)d_out, 0u, (long long)N_NODES * OUT_DIM);

    edge_softmax_max<<<cdiv((long long)(N_EDGES + N_NODES), 256), 256, 0, stream>>>(
        src, dst, AS2, AD2, M2, N_EDGES, N_NODES, 1);
    edge_softmax_acc<<<cdiv((long long)(N_EDGES + N_NODES) * 32, 256), 256, 0, stream>>>(
        src, dst, AS2, AD2, M2, HP2, DN2, (float*)d_out, N_EDGES, N_NODES, 1, OUT_DIM);
    gat_finalize<<<cdiv((long long)N_NODES * OUT_DIM, 256), 256, 0, stream>>>(
        (float*)d_out, DN2, bias2, N_NODES, 1, OUT_DIM, /*relu=*/0);

    (void)in_sizes; (void)n_in; (void)out_size; (void)ws_size;
}
